// OneHeadAttention_60928406061153
// MI455X (gfx1250) — matrix operand order repaired and checked by reading, not compiled, not run on hardware
//
#include <hip/hip_runtime.h>

// ---------------------------------------------------------------------------
// One-head causal attention for MI455X (gfx1250), wave32 + WMMA f16.
//   Phase 0: convert Wq/Wk/Wv f32 -> f16 once (L2-resident).
//   Phase 1: Q/K/V projections via v_wmma_f32_16x16x32_f16.  Q stored
//            pre-scaled by 1/sqrt(64).  Q,K row-major f16 [B*T,64];
//            V transposed f16 [B][64][T] so phase-2 A-frag loads are b128.
//   Phase 2: flash attention.  S^T = K*Q^T so the softmax reduction is
//            per-lane + one shfl_xor(16); O^T = V^T*P^T accumulated in f32.
//            All 16 b128 chunk loads issued up-front so they overlap the
//            S^T WMMAs + softmax VALU.  Mask only on the diagonal chunk.
//            Long (high-q0) tiles scheduled first for tail balance.
// ---------------------------------------------------------------------------

typedef __attribute__((ext_vector_type(16))) _Float16 v16h;
typedef __attribute__((ext_vector_type(8)))  float    v8f;

#define TSEQ 4096
#define DEMB 512
#define DK   64
#define NB   4
#define TOT  (NB * TSEQ)
#define WELEM (DK * DEMB)   // 32768 elements per weight matrix

union Frag {
  v16h v;
  _Float16 h[16];
  uint4 q[2];
};

static __device__ __forceinline__ v8f wmma16(const Frag& a, const Frag& b, v8f c) {
  return __builtin_amdgcn_wmma_f32_16x16x32_f16(
      /*neg_a=*/false, a.v, /*neg_b=*/false, b.v,
      /*c_mod=*/(short)0, c, /*reuse_a=*/false, /*reuse_b=*/false);
}

// ---------------------------------------------------------------------------
// Phase 0: one-shot f32 -> f16 conversion of the three weight matrices.
// ---------------------------------------------------------------------------
__global__ __launch_bounds__(256) void convert_w_kernel(
    const float* __restrict__ Wq, const float* __restrict__ Wk,
    const float* __restrict__ Wv, _Float16* __restrict__ Wh)
{
  const int i = blockIdx.x * 256 + threadIdx.x;   // 0 .. 3*WELEM-1
  const int m = i / WELEM;
  const int r = i - m * WELEM;
  const float* src = (m == 0) ? Wq : (m == 1) ? Wk : Wv;
  Wh[i] = (_Float16)src[r];
}

// ---------------------------------------------------------------------------
// Phase 1: projections.  One wave per 16-row tile of X (1024 tiles).
// ---------------------------------------------------------------------------
__global__ __launch_bounds__(128) void proj_qkv_kernel(
    const float* __restrict__ X, const _Float16* __restrict__ Wh,
    _Float16* __restrict__ Qw, _Float16* __restrict__ Kw,
    _Float16* __restrict__ Vt)
{
  const int lane = threadIdx.x & 31;
  const int wid  = threadIdx.x >> 5;
  const int tile = blockIdx.x * 4 + wid;
  const int lo = lane & 15;
  const int hi = lane >> 4;
  const int row0 = tile * 16;

  v8f acc[3][4];
#pragma unroll
  for (int m = 0; m < 3; ++m)
#pragma unroll
    for (int n = 0; n < 4; ++n)
      acc[m][n] = (v8f){0.f, 0.f, 0.f, 0.f, 0.f, 0.f, 0.f, 0.f};

  const float* xrow = X + (size_t)(row0 + lo) * DEMB;

  for (int es = 0; es < DEMB / 32; ++es) {
    const int e0 = es * 32;
    // A fragment (X tile, 16x32 f16): lane=row, halves h -> e = e0+8*hi+(h&7)+16*(h>>3)
    Frag a;
    {
      const float* p = xrow + e0 + 8 * hi;
#pragma unroll
      for (int h0 = 0; h0 < 8; ++h0) a.h[h0]     = (_Float16)p[h0];
#pragma unroll
      for (int h0 = 0; h0 < 8; ++h0) a.h[8 + h0] = (_Float16)p[16 + h0];
    }
#pragma unroll
    for (int m = 0; m < 3; ++m) {
#pragma unroll
      for (int nt = 0; nt < 4; ++nt) {
        // B fragment (W^T tile, 32x16): lane=col k, halves h -> e = e0+8*hi+(h&7)+16*(h>>3)
        Frag bfr;
        const _Float16* wp =
            Wh + (size_t)m * WELEM + (size_t)(nt * 16 + lo) * DEMB + e0 + 8 * hi;
        bfr.q[0] = *(const uint4*)(wp);
        bfr.q[1] = *(const uint4*)(wp + 16);
        acc[m][nt] = wmma16(a, bfr, acc[m][nt]);
      }
    }
  }

  // D layout: lane holds col k = nt*16+lo; VGPR r -> row t = row0 + r + 8*hi.
  const float qscale = 0.125f;  // 1/sqrt(DK), folded into Q
#pragma unroll
  for (int nt = 0; nt < 4; ++nt) {
    const int k = nt * 16 + lo;
#pragma unroll
    for (int r = 0; r < 8; ++r) {
      const size_t t = (size_t)(row0 + r + 8 * hi);
      Qw[t * DK + k] = (_Float16)(acc[0][nt][r] * qscale);
      Kw[t * DK + k] = (_Float16)acc[1][nt][r];
    }
  }
  // V transposed: [b][64][TSEQ] f16 -> 8 consecutive t per lane = one b128 store.
  const int bb = row0 / TSEQ;
  const int tl = row0 % TSEQ;
#pragma unroll
  for (int nt = 0; nt < 4; ++nt) {
    const int k = nt * 16 + lo;
    Frag pk;
#pragma unroll
    for (int r = 0; r < 8; ++r) pk.h[r] = (_Float16)acc[2][nt][r];
    *(uint4*)(Vt + (size_t)bb * DK * TSEQ + (size_t)k * TSEQ + tl + 8 * hi) = pk.q[0];
  }
}

// ---------------------------------------------------------------------------
// Phase 2 helper: process one 32-key chunk (MASK only for the diagonal chunk).
// ---------------------------------------------------------------------------
template <bool MASK>
static __device__ __forceinline__ void process_chunk(
    const int t0, const int qg,
    const _Float16* __restrict__ kbase, const _Float16* __restrict__ vbase,
    const Frag qb[2], v8f o[4], float& mrun, float& srun,
    const int lo, const int hi)
{
  // --- issue ALL chunk loads up front (16 x b128) so they overlap compute ---
  Frag ka[2][2];  // K A-frags: lane=row t, halves h -> kd = 8*hi+(h&7)+16*(h>>3)
#pragma unroll
  for (int tm = 0; tm < 2; ++tm) {
    const _Float16* kp = kbase + (size_t)(t0 + tm * 16 + lo) * DK;
    ka[tm][0].q[0] = *(const uint4*)(kp + 8 * hi);
    ka[tm][0].q[1] = *(const uint4*)(kp + 8 * hi + 16);
    ka[tm][1].q[0] = *(const uint4*)(kp + 32 + 8 * hi);
    ka[tm][1].q[1] = *(const uint4*)(kp + 32 + 8 * hi + 16);
  }
  Frag va[4];     // V^T A-frags: lane=row k, halves -> t (contiguous)
#pragma unroll
  for (int mt = 0; mt < 4; ++mt) {
    const _Float16* vp = vbase + (size_t)(mt * 16 + lo) * TSEQ + t0;
    va[mt].q[0] = *(const uint4*)(vp + 16 * hi);
    va[mt].q[1] = *(const uint4*)(vp + 16 * hi + 8);
  }

  // --- S^T tiles: st[tm] = (K rows t0+16*tm..+15) x Q^T  [16t x 16q] ---
  v8f st[2];
#pragma unroll
  for (int tm = 0; tm < 2; ++tm) {
    v8f c = (v8f){0.f, 0.f, 0.f, 0.f, 0.f, 0.f, 0.f, 0.f};
    c = wmma16(ka[tm][0], qb[0], c);
    c = wmma16(ka[tm][1], qb[1], c);
    st[tm] = c;
  }

  // --- causal mask (diagonal chunk only); element t = t0 + 16*tm + 8*hi + r ---
  if (MASK) {
#pragma unroll
    for (int tm = 0; tm < 2; ++tm)
#pragma unroll
      for (int r = 0; r < 8; ++r)
        if ((t0 + tm * 16 + 8 * hi + r) > qg) st[tm][r] = -3.0e38f;
  }

  // --- online softmax (per-lane + one half-wave exchange) ---
  float lm = -3.0e38f;
#pragma unroll
  for (int r = 0; r < 8; ++r) lm = fmaxf(lm, fmaxf(st[0][r], st[1][r]));
  lm = fmaxf(lm, __shfl_xor(lm, 16, 32));
  const float mnew = fmaxf(mrun, lm);
  const float corr = __expf(mrun - mnew);

  float p0[8], p1[8];
  float ls = 0.0f;
#pragma unroll
  for (int r = 0; r < 8; ++r) {
    p0[r] = __expf(st[0][r] - mnew);
    p1[r] = __expf(st[1][r] - mnew);
    ls += p0[r] + p1[r];
  }
  ls += __shfl_xor(ls, 16, 32);
  srun = srun * corr + ls;
  mrun = mnew;

#pragma unroll
  for (int mt = 0; mt < 4; ++mt)
#pragma unroll
    for (int r = 0; r < 8; ++r) o[mt][r] *= corr;

  // --- build P^T B-fragment [32t x 16q]: halves h -> t = t0 + 16*hi + h ---
  Frag pb;
#pragma unroll
  for (int r = 0; r < 8; ++r) {
    const float x0 = __shfl_xor(p0[r], 16, 32);  // partner's st0 exp
    const float x1 = __shfl_xor(p1[r], 16, 32);  // partner's st1 exp
    pb.h[r]     = (_Float16)(hi ? x1 : p0[r]);   // t = t0 + 16*hi + r
    pb.h[8 + r] = (_Float16)(hi ? p1[r] : x0);   // t = t0 + 16*hi + 8 + r
  }

  // --- O^T += V^T * P^T ---
#pragma unroll
  for (int mt = 0; mt < 4; ++mt)
    o[mt] = wmma16(va[mt], pb, o[mt]);
}

// ---------------------------------------------------------------------------
// Phase 2: flash attention.  One wave per 16-query tile, 32-key chunks.
// ---------------------------------------------------------------------------
__global__ __launch_bounds__(128) void attn_fwd_kernel(
    const _Float16* __restrict__ Qw, const _Float16* __restrict__ Kw,
    const _Float16* __restrict__ Vt, float* __restrict__ Out)
{
  const int lane = threadIdx.x & 31;
  const int wid  = threadIdx.x >> 5;
  const int tid  = blockIdx.x * 4 + wid;       // 1024 query tiles
  const int lo = lane & 15;
  const int hi = lane >> 4;
  const int bb = tid >> 8;                     // 256 tiles per batch
  const int q0 = (255 - (tid & 255)) * 16;     // long tiles first (tail balance)
  const int qg = q0 + lo;                      // my query column (within batch)
  const size_t rowbase = (size_t)bb * TSEQ;

  // Q^T B-fragments: lane=col q, halves h -> kd = 32*c + 8*hi + (h&7) + 16*(h>>3).
  Frag qb[2];
  {
    const _Float16* qp = Qw + (rowbase + q0 + lo) * DK;
    qb[0].q[0] = *(const uint4*)(qp + 8 * hi);
    qb[0].q[1] = *(const uint4*)(qp + 8 * hi + 16);
    qb[1].q[0] = *(const uint4*)(qp + 32 + 8 * hi);
    qb[1].q[1] = *(const uint4*)(qp + 32 + 8 * hi + 16);
  }

  v8f o[4];
#pragma unroll
  for (int mt = 0; mt < 4; ++mt)
    o[mt] = (v8f){0.f, 0.f, 0.f, 0.f, 0.f, 0.f, 0.f, 0.f};

  float mrun = -3.0e38f;
  float srun = 0.0f;

  const _Float16* kbase = Kw + rowbase * DK;
  const _Float16* vbase = Vt + (size_t)bb * DK * TSEQ;

  // Exactly one chunk (the last) straddles the diagonal and needs masking.
  const int t_diag = ((q0 + 15) >> 5) << 5;

  for (int t0 = 0; t0 < t_diag; t0 += 32)
    process_chunk<false>(t0, qg, kbase, vbase, qb, o, mrun, srun, lo, hi);
  process_chunk<true>(t_diag, qg, kbase, vbase, qb, o, mrun, srun, lo, hi);

  // --- epilogue: O^T D-layout: lane=col q, k = mt*16 + 8*hi + r (contiguous) ---
  const float inv = 1.0f / srun;
  float* orow = Out + (rowbase + (size_t)qg) * DK;
#pragma unroll
  for (int mt = 0; mt < 4; ++mt) {
    float4 fa, fb;
    fa.x = o[mt][0] * inv; fa.y = o[mt][1] * inv;
    fa.z = o[mt][2] * inv; fa.w = o[mt][3] * inv;
    fb.x = o[mt][4] * inv; fb.y = o[mt][5] * inv;
    fb.z = o[mt][6] * inv; fb.w = o[mt][7] * inv;
    *(float4*)(orow + mt * 16 + 8 * hi)     = fa;
    *(float4*)(orow + mt * 16 + 8 * hi + 4) = fb;
  }
}

// ---------------------------------------------------------------------------
extern "C" void kernel_launch(void* const* d_in, const int* in_sizes, int n_in,
                              void* d_out, int out_size, void* d_ws, size_t ws_size,
                              hipStream_t stream) {
  const float* X  = (const float*)d_in[0];
  const float* Wq = (const float*)d_in[1];
  const float* Wk = (const float*)d_in[2];
  const float* Wv = (const float*)d_in[3];

  _Float16* Qw = (_Float16*)d_ws;             // [TOT][64]    f16  (2 MB)
  _Float16* Kw = Qw + (size_t)TOT * DK;       // [TOT][64]    f16  (2 MB)
  _Float16* Vt = Kw + (size_t)TOT * DK;       // [B][64][T]   f16  (2 MB)
  _Float16* Wh = Vt + (size_t)NB * DK * TSEQ; // [3][64][512] f16  (192 KB)

  convert_w_kernel<<<(3 * WELEM) / 256, 256, 0, stream>>>(Wq, Wk, Wv, Wh);

  const int tiles = TOT / 16;                 // 1024 query/row tiles
  proj_qkv_kernel<<<tiles / 4, 128, 0, stream>>>(X, Wh, Qw, Kw, Vt);
  attn_fwd_kernel<<<tiles / 4, 128, 0, stream>>>(Qw, Kw, Vt, (float*)d_out);
}
